// ViT_76398878261801
// MI455X (gfx1250) — compile-verified, matrix operand order checked by reading, not run on hardware
//
#include <hip/hip_runtime.h>
#include <math.h>

// ---------------- model constants ----------------
#define L_    12
#define D_    768
#define H_    12
#define DH_   64
#define F_    3072
#define P_    16
#define IMG_  384
#define BB_   16
#define NC_   1000
#define S_    577
#define NP_   576          // patches per image
#define BS_   (BB_ * S_)   // 9232 rows
#define MP_   (BB_ * NP_)  // 9216 patch rows

typedef __bf16 v16bf __attribute__((ext_vector_type(16)));
typedef __bf16 v8bf  __attribute__((ext_vector_type(8)));
typedef __bf16 v4bf  __attribute__((ext_vector_type(4)));
typedef float  v8f   __attribute__((ext_vector_type(8)));
typedef float  v4f   __attribute__((ext_vector_type(4)));
typedef int    v4i   __attribute__((ext_vector_type(4)));

#if defined(__gfx1250__) && __has_builtin(__builtin_amdgcn_global_load_async_to_lds_b128)
#define HAS_ASYNC_LDS 1
#else
#define HAS_ASYNC_LDS 0
#endif

#define AS1_ __attribute__((address_space(1)))
#define AS3_ __attribute__((address_space(3)))

static __device__ __forceinline__ v16bf cat8(v8bf lo, v8bf hi) {
    return __builtin_shufflevector(lo, hi, 0,1,2,3,4,5,6,7,8,9,10,11,12,13,14,15);
}

static __device__ __forceinline__ v4bf cvt4(v4f v) {
    v4bf r;
    r.x = (__bf16)v.x; r.y = (__bf16)v.y; r.z = (__bf16)v.z; r.w = (__bf16)v.w;
    return r;
}

static __device__ __forceinline__ float gelu_tanh(float x) {
    float x3 = x * x * x;
    return 0.5f * x * (1.0f + tanhf(0.7978845608028654f * (x + 0.044715f * x3)));
}

// =====================================================================
// im2col (float4 -> bf16x4): x[B,3,384,384] -> patches[MP_][768] bf16
// =====================================================================
__global__ __launch_bounds__(256) void im2col_kernel(const float* __restrict__ x,
                                                     __bf16* __restrict__ out) {
    int idx = blockIdx.x * 256 + threadIdx.x;   // f4 index
    if (idx >= MP_ * 192) return;
    int k4  = idx % 192;
    int row = idx / 192;
    int k = k4 * 4;
    int b = row / NP_, p = row % NP_;
    int py = p / 24, px = p % 24;
    int c = k / 256, rem = k % 256;
    int i = rem / 16, j = rem % 16;
    v4f v = *(const v4f*)(x + ((size_t)(b * 3 + c) * IMG_ + (py * 16 + i)) * IMG_ + px * 16 + j);
    *(v4bf*)(out + (size_t)row * D_ + k) = cvt4(v);
}

// =====================================================================
// assemble h[b,s,:] = (s==0 ? cls : patch_out[b*576+s-1]) + pos[s]  (f32)
// =====================================================================
__global__ __launch_bounds__(256) void embed_kernel(const float* __restrict__ tmp,
                                                    const float* __restrict__ cls,
                                                    const float* __restrict__ pos,
                                                    float* __restrict__ h) {
    int idx = blockIdx.x * 256 + threadIdx.x;   // f4 index
    if (idx >= BS_ * 192) return;
    int d4 = idx % 192;
    int bs = idx / 192;
    int s = bs % S_, b = bs / S_;
    int d = d4 * 4;
    v4f v = (s == 0) ? *(const v4f*)(cls + d)
                     : *(const v4f*)(tmp + (size_t)(b * NP_ + s - 1) * D_ + d);
    v4f pe = *(const v4f*)(pos + (size_t)s * D_ + d);
    *(v4f*)(h + (size_t)bs * D_ + d) = v + pe;
}

// =====================================================================
// LayerNorm: one block per f32 row of 768 -> bf16 output row
// =====================================================================
__global__ __launch_bounds__(256) void ln_kernel(const float* __restrict__ x,
                                                 const float* __restrict__ g,
                                                 const float* __restrict__ bb,
                                                 __bf16* __restrict__ y) {
    __shared__ float red[256];
    int row = blockIdx.x;
    int tid = threadIdx.x;
    const float* xr = x + (size_t)row * D_;
    float v0 = xr[tid], v1 = xr[tid + 256], v2 = xr[tid + 512];
    red[tid] = v0 + v1 + v2;
    __syncthreads();
    for (int st = 128; st > 0; st >>= 1) {
        if (tid < st) red[tid] += red[tid + st];
        __syncthreads();
    }
    float mean = red[0] * (1.0f / 768.0f);
    __syncthreads();
    float d0 = v0 - mean, d1 = v1 - mean, d2 = v2 - mean;
    red[tid] = d0 * d0 + d1 * d1 + d2 * d2;
    __syncthreads();
    for (int st = 128; st > 0; st >>= 1) {
        if (tid < st) red[tid] += red[tid + st];
        __syncthreads();
    }
    float rstd = rsqrtf(red[0] * (1.0f / 768.0f) + 1e-6f);
    __bf16* yr = y + (size_t)row * D_;
    yr[tid]       = (__bf16)(d0 * rstd * g[tid]       + bb[tid]);
    yr[tid + 256] = (__bf16)(d1 * rstd * g[tid + 256] + bb[tid + 256]);
    yr[tid + 512] = (__bf16)(d2 * rstd * g[tid + 512] + bb[tid + 512]);
}

// =====================================================================
// bf16-WMMA GEMM:  C = op(A(bf16)[M,K] @ B(f32)[K,N] + bias) (+resid)
//   128x128 tile, BK=32, 256 threads = 8 waves, wave => 32x64 (2x4 WMMA)
//   Double-buffered LDS; A staged as raw 16-B bf16 copies; B converted.
// =====================================================================
template <int TRANSB, int OUTBF16>
__global__ __launch_bounds__(256) void gemm_bf16_kernel(
    const __bf16* __restrict__ A, int lda,
    const float* __restrict__ Bw, int ldb,
    const float* __restrict__ bias,
    const float* __restrict__ resid,
    float* __restrict__ C, __bf16* __restrict__ Cb, int ldc,
    int M, int N, int K, int gelu_flag) {

    __shared__ alignas(16) __bf16 lsA[2][128][40];   // [m][k], padded
    __shared__ alignas(16) __bf16 lsB[2][128][40];   // [n][k] (transposed), padded

    int tid  = threadIdx.x;
    int wave = tid >> 5, lane = tid & 31;
    int wr = wave >> 1, wc = wave & 1;               // 4x2 wave grid
    int bm0 = blockIdx.y * 128, bn0 = blockIdx.x * 128;
    int half = lane >> 4, l15 = lane & 15;

    v8f acc[2][4];
    #pragma unroll
    for (int mi = 0; mi < 2; mi++)
        #pragma unroll
        for (int ni = 0; ni < 4; ni++)
            acc[mi][ni] = (v8f)(0.0f);

    v8bf aReg[2];
    v4f  bReg[4];

    auto loadA = [&](int k0) {
        #pragma unroll
        for (int i = 0; i < 2; i++) {
            int e = tid + 256 * i;
            int r = e >> 2, c8 = e & 3;
            int gr = bm0 + r;
            int grc = gr < M ? gr : (M - 1);
            aReg[i] = *(const v8bf*)(A + (size_t)grc * lda + k0 + c8 * 8);
        }
    };
    auto storeA = [&](int buf) {
        #pragma unroll
        for (int i = 0; i < 2; i++) {
            int e = tid + 256 * i;
            int r = e >> 2, c8 = e & 3;
            *(v8bf*)&lsA[buf][r][c8 * 8] = aReg[i];
        }
    };
    auto loadB = [&](int k0) {
        if (TRANSB) {
            #pragma unroll
            for (int i = 0; i < 4; i++) {
                int e = tid + 256 * i;
                int n = e >> 3, c4 = e & 7;
                int gn = bn0 + n;
                int gnc = gn < N ? gn : (N - 1);
                bReg[i] = *(const v4f*)(Bw + (size_t)gnc * ldb + k0 + c4 * 4);
            }
        } else {
            #pragma unroll
            for (int i = 0; i < 4; i++) {
                int e = tid + 256 * i;
                int n = e & 127, k4 = e >> 7;
                int gn = bn0 + n;
                int gnc = gn < N ? gn : (N - 1);
                v4f v;
                #pragma unroll
                for (int t = 0; t < 4; t++)
                    v[t] = Bw[(size_t)(k0 + k4 * 4 + t) * ldb + gnc];
                bReg[i] = v;
            }
        }
    };
    auto storeB = [&](int buf) {
        if (TRANSB) {
            #pragma unroll
            for (int i = 0; i < 4; i++) {
                int e = tid + 256 * i;
                int n = e >> 3, c4 = e & 7;
                *(v4bf*)&lsB[buf][n][c4 * 4] = cvt4(bReg[i]);
            }
        } else {
            #pragma unroll
            for (int i = 0; i < 4; i++) {
                int e = tid + 256 * i;
                int n = e & 127, k4 = e >> 7;
                *(v4bf*)&lsB[buf][n][k4 * 4] = cvt4(bReg[i]);
            }
        }
    };
    auto compute = [&](int buf) {
        v16bf af[2], bf_[4];
        #pragma unroll
        for (int mi = 0; mi < 2; mi++) {
            int r = wr * 32 + mi * 16 + l15;
            v8bf lo = *(const v8bf*)&lsA[buf][r][half * 8];
            v8bf hi = *(const v8bf*)&lsA[buf][r][16 + half * 8];
            af[mi] = cat8(lo, hi);
        }
        #pragma unroll
        for (int ni = 0; ni < 4; ni++) {
            int n = wc * 64 + ni * 16 + l15;
            v8bf lo = *(const v8bf*)&lsB[buf][n][half * 8];
            v8bf hi = *(const v8bf*)&lsB[buf][n][16 + half * 8];
            bf_[ni] = cat8(lo, hi);
        }
        #pragma unroll
        for (int mi = 0; mi < 2; mi++)
            #pragma unroll
            for (int ni = 0; ni < 4; ni++)
                acc[mi][ni] = __builtin_amdgcn_wmma_f32_16x16x32_bf16(
                    false, af[mi], false, bf_[ni], (short)0, acc[mi][ni], false, false);
    };

    // ---- software pipeline: double-buffered LDS ----
    int nk = K >> 5;
    loadA(0); loadB(0);
    storeA(0); storeB(0);
    for (int it = 0; it < nk; it++) {
        int cur = it & 1;
        __syncthreads();                       // lds[cur] visible
        if (it + 1 < nk) { loadA((it + 1) << 5); loadB((it + 1) << 5); }
        compute(cur);
        if (it + 1 < nk) {
            __syncthreads();                   // all waves done reading lds[cur^1]
            storeA(cur ^ 1); storeB(cur ^ 1);
        }
    }

    // ---- epilogue: bias -> gelu -> residual -> store ----
    #pragma unroll
    for (int mi = 0; mi < 2; mi++) {
        #pragma unroll
        for (int ni = 0; ni < 4; ni++) {
            int col = bn0 + wc * 64 + ni * 16 + l15;
            if (col >= N) continue;
            float bval = bias ? bias[col] : 0.0f;
            #pragma unroll
            for (int r = 0; r < 8; r++) {
                int row = bm0 + wr * 32 + mi * 16 + r + 8 * half;
                if (row >= M) continue;
                float v = acc[mi][ni][r] + bval;
                if (gelu_flag) v = gelu_tanh(v);
                if (OUTBF16) {
                    Cb[(size_t)row * ldc + col] = (__bf16)v;
                } else {
                    if (resid) v += resid[(size_t)row * ldc + col];
                    C[(size_t)row * ldc + col] = v;
                }
            }
        }
    }
}

// =====================================================================
// Flash attention (bf16 in / bf16 out): grid (ceil(S/128), H, B),
// 256 threads = 8 waves, each wave owns 16 query rows;
// K/V streamed in 64-key LDS blocks (K via async-to-LDS when available).
// =====================================================================
__global__ __launch_bounds__(256) void attn_kernel(const __bf16* __restrict__ Qb,
                                                   const __bf16* __restrict__ Kb,
                                                   const __bf16* __restrict__ Vb,
                                                   __bf16* __restrict__ Ob) {
    __shared__ alignas(16) __bf16 lsK[64][72];      // [key][dim]
    __shared__ alignas(16) __bf16 lsV[64][72];      // [dim][key]
    __shared__ float  lsS[8][16][64];               // per-wave scores
    __shared__ alignas(16) __bf16 lsP[8][16][72];   // per-wave probs (bf16)
    __shared__ float  lsAl[8][16];
    __shared__ float  lsL[8][16];

    int tid = threadIdx.x;
    int wave = tid >> 5, lane = tid & 31;
    int half = lane >> 4, l15 = lane & 15;
    int h = blockIdx.y, b = blockIdx.z;
    int q0 = blockIdx.x * 128 + wave * 16;
    const float scale = 0.125f;  // 1/sqrt(64), applied in softmax domain

    // ---- load Q fragments (raw bf16, no conversion) ----
    v16bf qf[2];
    {
        int qrow = q0 + l15;
        int qrc = qrow < S_ ? qrow : (S_ - 1);
        const __bf16* qbase = Qb + ((size_t)(b * S_ + qrc)) * D_ + h * DH_;
        #pragma unroll
        for (int kc = 0; kc < 2; kc++) {
            v8bf lo = *(const v8bf*)(qbase + kc * 32 + half * 8);
            v8bf hi = *(const v8bf*)(qbase + kc * 32 + 16 + half * 8);
            qf[kc] = cat8(lo, hi);
        }
    }

    v8f oacc[4];
    #pragma unroll
    for (int t = 0; t < 4; t++) oacc[t] = (v8f)(0.0f);
    float m = -1e30f, l = 0.0f;

    for (int j0 = 0; j0 < S_; j0 += 64) {
        int nvalid = S_ - j0;
        if (nvalid > 64) nvalid = 64;
        __syncthreads();

        // ---- K block: 128 key rows x 32 dims, identity bf16 copies ----
#if HAS_ASYNC_LDS
        #pragma unroll
        for (int i = 0; i < 2; i++) {
            int e = tid + 256 * i;
            int key = e >> 2, dm8 = e & 3;
            int kc_ = key < nvalid ? key : 0;
            const __bf16* gp = Kb + ((size_t)(b * S_ + j0 + kc_)) * D_ + h * DH_ + dm8 * 8;
            __builtin_amdgcn_global_load_async_to_lds_b128(
                (AS1_ v4i*)(unsigned long long)gp,
                (AS3_ v4i*)(unsigned)(unsigned long long)&lsK[key][dm8 * 8],
                0, 0);
        }
#else
        #pragma unroll
        for (int i = 0; i < 2; i++) {
            int e = tid + 256 * i;
            int key = e >> 2, dm8 = e & 3;
            int kc_ = key < nvalid ? key : 0;
            const __bf16* gp = Kb + ((size_t)(b * S_ + j0 + kc_)) * D_ + h * DH_ + dm8 * 8;
            *(v8bf*)&lsK[key][dm8 * 8] = *(const v8bf*)gp;
        }
#endif
        // ---- V block: transposed stage; coalesced strided bf16 loads ----
        v4bf vv[4];
        #pragma unroll
        for (int i = 0; i < 4; i++) {
            int e = tid + 256 * i;
            int dm = e & 63, key4 = e >> 6;
            v4bf v;
            #pragma unroll
            for (int t = 0; t < 4; t++) {
                int key = key4 * 4 + t;
                int kc_ = key < nvalid ? key : 0;
                v[t] = Vb[((size_t)(b * S_ + j0 + kc_)) * D_ + h * DH_ + dm];
            }
            vv[i] = v;
        }
        #pragma unroll
        for (int i = 0; i < 4; i++) {
            int e = tid + 256 * i;
            int dm = e & 63, key4 = e >> 6;
            *(v4bf*)&lsV[dm][key4 * 4] = vv[i];
        }
#if HAS_ASYNC_LDS
#if __has_builtin(__builtin_amdgcn_s_wait_asynccnt)
        __builtin_amdgcn_s_wait_asynccnt(0);
#else
        asm volatile("s_wait_asynccnt 0" ::: "memory");
#endif
#endif
        __syncthreads();

        // ---- scores = Q @ K^T : 4 column tiles x 2 K-chunks ----
        v8f sc[4];
        #pragma unroll
        for (int t = 0; t < 4; t++) sc[t] = (v8f)(0.0f);
        #pragma unroll
        for (int kc = 0; kc < 2; kc++) {
            #pragma unroll
            for (int t = 0; t < 4; t++) {
                int n = t * 16 + l15;   // local key index
                v8bf lo = *(const v8bf*)&lsK[n][kc * 32 + half * 8];
                v8bf hi = *(const v8bf*)&lsK[n][kc * 32 + 16 + half * 8];
                sc[t] = __builtin_amdgcn_wmma_f32_16x16x32_bf16(
                    false, qf[kc], false, cat8(lo, hi), (short)0, sc[t], false, false);
            }
        }
        #pragma unroll
        for (int t = 0; t < 4; t++)
            #pragma unroll
            for (int r = 0; r < 8; r++)
                lsS[wave][r + 8 * half][t * 16 + l15] = sc[t][r];
        __syncthreads();

        // ---- online softmax: lane (row=l15) covers cols [half*32, +32) ----
        {
            int row = l15;
            float mx = m;
            for (int c = 0; c < 32; c++) {
                int cc = half * 32 + c;
                if (cc < nvalid) mx = fmaxf(mx, lsS[wave][row][cc] * scale);
            }
            mx = fmaxf(mx, __shfl_xor(mx, 16, 32));
            float alpha = __expf(m - mx);
            float sum = 0.0f;
            for (int c = 0; c < 32; c++) {
                int cc = half * 32 + c;
                float p = (cc < nvalid) ? __expf(lsS[wave][row][cc] * scale - mx) : 0.0f;
                lsP[wave][row][cc] = (__bf16)p;
                sum += p;
            }
            sum += __shfl_xor(sum, 16, 32);
            l = l * alpha + sum;
            m = mx;
            if (lane < 16) lsAl[wave][row] = alpha;
        }
        __syncthreads();

        // ---- rescale accumulators, then O += P @ V ----
        float a8[8];
        #pragma unroll
        for (int r = 0; r < 8; r++) a8[r] = lsAl[wave][r + 8 * half];
        #pragma unroll
        for (int t = 0; t < 4; t++)
            #pragma unroll
            for (int r = 0; r < 8; r++) oacc[t][r] *= a8[r];

        #pragma unroll
        for (int kc = 0; kc < 2; kc++) {
            v8bf plo = *(const v8bf*)&lsP[wave][l15][kc * 32 + half * 8];
            v8bf phi = *(const v8bf*)&lsP[wave][l15][kc * 32 + 16 + half * 8];
            v16bf pf = cat8(plo, phi);
            #pragma unroll
            for (int t = 0; t < 4; t++) {
                int n = t * 16 + l15;   // output dim
                v8bf vlo = *(const v8bf*)&lsV[n][kc * 32 + half * 8];
                v8bf vhi = *(const v8bf*)&lsV[n][kc * 32 + 16 + half * 8];
                oacc[t] = __builtin_amdgcn_wmma_f32_16x16x32_bf16(
                    false, pf, false, cat8(vlo, vhi), (short)0, oacc[t], false, false);
            }
        }
    }

    // ---- finalize: divide by l, store bf16 ----
    if (lane < 16) lsL[wave][lane] = l;
    __syncthreads();
    float li[8];
    #pragma unroll
    for (int r = 0; r < 8; r++) li[r] = 1.0f / lsL[wave][r + 8 * half];
    #pragma unroll
    for (int t = 0; t < 4; t++) {
        #pragma unroll
        for (int r = 0; r < 8; r++) {
            int qg = q0 + r + 8 * half;
            if (qg < S_)
                Ob[((size_t)(b * S_ + qg)) * D_ + h * DH_ + t * 16 + l15] =
                    (__bf16)(oacc[t][r] * li[r]);
        }
    }
}

// =====================================================================
// host orchestration
// =====================================================================
static inline dim3 gemm_grid(int M, int N) { return dim3((N + 127) / 128, (M + 127) / 128); }

extern "C" void kernel_launch(void* const* d_in, const int* in_sizes, int n_in,
                              void* d_out, int out_size, void* d_ws, size_t ws_size,
                              hipStream_t stream) {
    const float* x      = (const float*)d_in[0];
    const float* conv_w = (const float*)d_in[1];
    const float* conv_b = (const float*)d_in[2];
    const float* cls    = (const float*)d_in[3];
    const float* pos    = (const float*)d_in[4];
    const float* Wq = (const float*)d_in[5],  *bq = (const float*)d_in[6];
    const float* Wk = (const float*)d_in[7],  *bk = (const float*)d_in[8];
    const float* Wv = (const float*)d_in[9],  *bv = (const float*)d_in[10];
    const float* Wo = (const float*)d_in[11], *bo = (const float*)d_in[12];
    const float* W1 = (const float*)d_in[13], *b1 = (const float*)d_in[14];
    const float* W2 = (const float*)d_in[15], *b2 = (const float*)d_in[16];
    const float* ln1g = (const float*)d_in[17], *ln1b = (const float*)d_in[18];
    const float* ln2g = (const float*)d_in[19], *ln2b = (const float*)d_in[20];
    const float* nfg  = (const float*)d_in[21], *nfb  = (const float*)d_in[22];
    const float* fcw  = (const float*)d_in[23], *fcb  = (const float*)d_in[24];
    float* out = (float*)d_out;

    const size_t BSD = (size_t)BS_ * D_;  // 7,090,176 elements
    char* base = (char*)d_ws;
    float*  hbuf  = (float*)base;                                   // BSD f32
    float*  tmpf  = (float*)(base + BSD * 4);                       // patch GEMM out (f32)
    __bf16* ybf   = (__bf16*)(base + 2 * BSD * 4);                  // BSD bf16
    __bf16* qbf   = (__bf16*)(base + 2 * BSD * 4 + BSD * 2);        // 4*BSD bf16 region
    __bf16* kbf   = qbf + BSD;
    __bf16* vbf   = kbf + BSD;
    __bf16* obf   = vbf + BSD;
    __bf16* midbf = qbf;      // FFN intermediate: B*S*F == 4*BSD bf16
    __bf16* pbf   = ybf;      // im2col patches (MP_*D_ <= BSD)

    // 1) patch embedding: im2col(bf16) -> GEMM(conv_w^T) -> assemble
    {
        int tot = MP_ * 192;
        im2col_kernel<<<(tot + 255) / 256, 256, 0, stream>>>(x, pbf);
        gemm_bf16_kernel<1, 0><<<gemm_grid(MP_, D_), 256, 0, stream>>>(
            pbf, D_, conv_w, D_, conv_b, nullptr, tmpf, nullptr, D_, MP_, D_, D_, 0);
        int tot2 = BS_ * 192;
        embed_kernel<<<(tot2 + 255) / 256, 256, 0, stream>>>(tmpf, cls, pos, hbuf);
    }

    // 2) transformer layers
    for (int i = 0; i < L_; i++) {
        const float* wq = Wq + (size_t)i * D_ * D_;
        const float* wk = Wk + (size_t)i * D_ * D_;
        const float* wv = Wv + (size_t)i * D_ * D_;
        const float* wo = Wo + (size_t)i * D_ * D_;
        const float* w1 = W1 + (size_t)i * D_ * F_;
        const float* w2 = W2 + (size_t)i * F_ * D_;

        ln_kernel<<<BS_, 256, 0, stream>>>(hbuf, ln1g + i * D_, ln1b + i * D_, ybf);

        gemm_bf16_kernel<0, 1><<<gemm_grid(BS_, D_), 256, 0, stream>>>(
            ybf, D_, wq, D_, bq + i * D_, nullptr, nullptr, qbf, D_, BS_, D_, D_, 0);
        gemm_bf16_kernel<0, 1><<<gemm_grid(BS_, D_), 256, 0, stream>>>(
            ybf, D_, wk, D_, bk + i * D_, nullptr, nullptr, kbf, D_, BS_, D_, D_, 0);
        gemm_bf16_kernel<0, 1><<<gemm_grid(BS_, D_), 256, 0, stream>>>(
            ybf, D_, wv, D_, bv + i * D_, nullptr, nullptr, vbf, D_, BS_, D_, D_, 0);

        attn_kernel<<<dim3((S_ + 127) / 128, H_, BB_), 256, 0, stream>>>(
            qbf, kbf, vbf, obf);

        gemm_bf16_kernel<0, 0><<<gemm_grid(BS_, D_), 256, 0, stream>>>(
            obf, D_, wo, D_, bo + i * D_, hbuf, hbuf, nullptr, D_, BS_, D_, D_, 0);

        ln_kernel<<<BS_, 256, 0, stream>>>(hbuf, ln2g + i * D_, ln2b + i * D_, ybf);

        gemm_bf16_kernel<0, 1><<<gemm_grid(BS_, F_), 256, 0, stream>>>(
            ybf, D_, w1, F_, b1 + i * F_, nullptr, nullptr, midbf, F_, BS_, F_, D_, /*gelu=*/1);

        gemm_bf16_kernel<0, 0><<<gemm_grid(BS_, D_), 256, 0, stream>>>(
            midbf, F_, w2, D_, b2 + i * D_, hbuf, hbuf, nullptr, D_, BS_, D_, F_, 0);
    }

    // 3) final LN + classifier head on CLS rows (lda stride trick)
    ln_kernel<<<BS_, 256, 0, stream>>>(hbuf, nfg, nfb, ybf);
    gemm_bf16_kernel<0, 0><<<gemm_grid(BB_, NC_), 256, 0, stream>>>(
        ybf, S_ * D_, fcw, NC_, fcb, nullptr, out, nullptr, NC_, BB_, NC_, D_, 0);
}
